// TrajectoryAgent_2448131359396
// MI455X (gfx1250) — matrix operand order repaired and checked by reading, not compiled, not run on hardware
//
#include <hip/hip_runtime.h>
#include <hip/hip_bf16.h>
#include <math.h>

// Problem constants (from reference)
#define E_EDGES 200000
#define TOKD    256     // TOKEN_DIM == HIDDEN_DIM
#define ACT     512     // ACTION_DIM
#define MTILE   64      // edges per workgroup

typedef __attribute__((ext_vector_type(16))) __bf16 v16bf;
typedef __attribute__((ext_vector_type(8)))  float  v8f;

__device__ __forceinline__ v16bf mkfrag(uint4 lo, uint4 hi) {
  union { uint4 u[2]; v16bf v; } t;
  t.u[0] = lo; t.u[1] = hi;
  return t.v;
}

// Computes 4 M-tiles (16 rows each, 64 rows total) x one 16-col N-tile:
//   acc[mt] += A[mt*16..+16, 0..K) x W[nb..nb+16, 0..K)^T
// A: bf16 [64][K] row-major (LDS). W: bf16 [N][K] row-major (global, L2-resident).
// Fragment layouts used here (wave32; contiguous k order, same for A and B):
//   A 16x32 bf16: lane(l<16)=row l, holds K k0+0..15 contiguous; lane l+16 holds k0+16..31
//   B 32x16 bf16: lane(l<16)=col l, holds K k0+0..15 contiguous; lane l+16 holds k0+16..31
__device__ __forceinline__ void gemm_col4(const __bf16* __restrict__ A,
                                          const __bf16* __restrict__ W,
                                          int K, int nb, int lane, v8f acc[4]) {
  const int lrow = lane & 15;
  const int lh   = lane >> 4;
  const __bf16* wrow = W + (size_t)(nb + lrow) * K + lh * 16;
  for (int k0 = 0; k0 < K; k0 += 32) {
    v16bf bf = mkfrag(*(const uint4*)(wrow + k0), *(const uint4*)(wrow + k0 + 8));
#pragma unroll
    for (int mt = 0; mt < 4; ++mt) {
      const __bf16* ap = A + (size_t)(mt * 16 + lrow) * K + k0 + lh * 16;
      v16bf af = mkfrag(*(const uint4*)ap, *(const uint4*)(ap + 8));
      acc[mt] = __builtin_amdgcn_wmma_f32_16x16x32_bf16(
          /*neg_a=*/false, af, /*neg_b=*/false, bf,
          /*c_mod=*/(short)0, acc[mt], /*reuse_a=*/false, /*reuse_b=*/false);
    }
  }
}

// Prepass: convert weights fp32 -> bf16 into workspace (1.25 MB, L2-resident).
// ws layout (elements): Wr[65536] | Wn[65536] | W1[262144] | W2[262144]
__global__ void convert_weights(const float* __restrict__ Wr, const float* __restrict__ Wn,
                                const float* __restrict__ W1, const float* __restrict__ W2,
                                __bf16* __restrict__ ws) {
  int i = blockIdx.x * blockDim.x + threadIdx.x;
  if (i < 65536)       ws[i] = (__bf16)Wr[i];
  else if (i < 131072) ws[i] = (__bf16)Wn[i - 65536];
  else if (i < 393216) ws[i] = (__bf16)W1[i - 131072];
  else if (i < 655360) ws[i] = (__bf16)W2[i - 393216];
}

// Fused: gather tails, [rel@Wr^T | tail@Wn^T], GELU(.@W1^T + b1), .@W2^T + b2
__global__ void __launch_bounds__(256) fused_edge_mlp(
    const float* __restrict__ rel, const float* __restrict__ nodes,
    const int* __restrict__ eidx, const float* __restrict__ b1,
    const float* __restrict__ b2, const __bf16* __restrict__ wts,
    float* __restrict__ out) {
  // Region 0: stage0 holds rel[64][256] + tail[64][256]; reused as h[64][512] after layer 2.
  __shared__ __align__(16) __bf16 sIN[MTILE * ACT];   // 64 KB
  __shared__ __align__(16) __bf16 sAK[MTILE * ACT];   // 64 KB  (action_key, bf16)

  const int e0  = blockIdx.x * MTILE;
  const int tid = threadIdx.x;

  // ---- Stage 0: load + gather + fp32->bf16 into LDS ----
  {
    const int r  = tid >> 2;          // 0..63 : edge row within tile
    const int c0 = (tid & 3) << 6;    // 0,64,128,192
    const float* relp  = rel + (size_t)(e0 + r) * TOKD + c0;
    const int    tnode = eidx[E_EDGES + e0 + r];         // edge_index[1][e]
    const float* tailp = nodes + (size_t)tnode * TOKD + c0;
    __bf16* dR = sIN + r * TOKD + c0;
    __bf16* dT = sIN + MTILE * TOKD + r * TOKD + c0;
#pragma unroll
    for (int c = 0; c < 64; c += 4) {
      float4 v = *(const float4*)(relp + c);
      dR[c] = (__bf16)v.x; dR[c+1] = (__bf16)v.y; dR[c+2] = (__bf16)v.z; dR[c+3] = (__bf16)v.w;
      float4 u = *(const float4*)(tailp + c);
      dT[c] = (__bf16)u.x; dT[c+1] = (__bf16)u.y; dT[c+2] = (__bf16)u.z; dT[c+3] = (__bf16)u.w;
    }
  }
  __syncthreads();

  const int w = tid >> 5, lane = tid & 31;
  const int lrow = lane & 15, lh = lane >> 4;
  const __bf16* WrB = wts;
  const __bf16* WnB = wts + 65536;
  const __bf16* W1B = wts + 131072;
  const __bf16* W2B = wts + 393216;

  // ---- Layer 1: action_key = [rel@Wr^T | tail@Wn^T] -> sAK (bf16) ----
  {
    const __bf16* A = (w < 4) ? sIN : (sIN + MTILE * TOKD);
    const __bf16* W = (w < 4) ? WrB : WnB;
    const int nw0   = (w & 3) * 64;          // each half: 4 waves x 64 cols = 256
    const int cbase = (w < 4) ? 0 : 256;     // concat offset in action_key
    for (int nt = 0; nt < 4; ++nt) {
      const int nb = nw0 + nt * 16;
      v8f acc[4] = {};
      gemm_col4(A, W, TOKD, nb, lane, acc);
      const int col = cbase + nb + lrow;
#pragma unroll
      for (int mt = 0; mt < 4; ++mt)
#pragma unroll
        for (int r = 0; r < 8; ++r)
          sAK[(mt * 16 + lh * 8 + r) * ACT + col] = (__bf16)acc[mt][r];
    }
  }
  __syncthreads();

  // ---- Layer 2: h = GELU(action_key @ W1^T + b1) -> sH (= sIN region) ----
  __bf16* sH = sIN;
  {
    const int nw0 = w * 64;                  // 8 waves x 64 cols = 512
    for (int nt = 0; nt < 4; ++nt) {
      const int nb = nw0 + nt * 16;
      v8f acc[4] = {};
      gemm_col4(sAK, W1B, ACT, nb, lane, acc);
      const float bias = b1[nb + lrow];
#pragma unroll
      for (int mt = 0; mt < 4; ++mt)
#pragma unroll
        for (int r = 0; r < 8; ++r) {
          float x = acc[mt][r] + bias;
          float g = 0.5f * x * (1.0f + erff(x * 0.70710678118654752f)); // exact GELU
          sH[(mt * 16 + lh * 8 + r) * ACT + nb + lrow] = (__bf16)g;
        }
    }
  }
  __syncthreads();

  // ---- Layer 3: out = h @ W2^T + b2 -> global fp32 ----
  {
    const int nw0 = w * 64;
    for (int nt = 0; nt < 4; ++nt) {
      const int nb = nw0 + nt * 16;
      v8f acc[4] = {};
      gemm_col4(sH, W2B, ACT, nb, lane, acc);
      const float bias = b2[nb + lrow];
#pragma unroll
      for (int mt = 0; mt < 4; ++mt)
#pragma unroll
        for (int r = 0; r < 8; ++r)
          out[(size_t)(e0 + mt * 16 + lh * 8 + r) * ACT + nb + lrow] = acc[mt][r] + bias;
    }
  }
}

extern "C" void kernel_launch(void* const* d_in, const int* in_sizes, int n_in,
                              void* d_out, int out_size, void* d_ws, size_t ws_size,
                              hipStream_t stream) {
  const float* rel   = (const float*)d_in[0];  // relation_tokens [E,256]
  const float* nodes = (const float*)d_in[1];  // node_tokens [N,256]
  const int*   eidx  = (const int*)d_in[2];    // edge_index [2,E] int32
  const float* Wr    = (const float*)d_in[3];  // [256,256]
  const float* Wn    = (const float*)d_in[4];  // [256,256]
  const float* W1    = (const float*)d_in[5];  // [512,512]
  const float* b1    = (const float*)d_in[6];  // [512]
  const float* W2    = (const float*)d_in[7];  // [512,512]
  const float* b2    = (const float*)d_in[8];  // [512]
  float*  out = (float*)d_out;                 // [E,512] fp32
  __bf16* wts = (__bf16*)d_ws;                 // 1.31 MB of workspace used

  convert_weights<<<2560, 256, 0, stream>>>(Wr, Wn, W1, W2, wts);
  fused_edge_mlp<<<E_EDGES / MTILE, 256, 0, stream>>>(rel, nodes, eidx, b1, b2, wts, out);
}
